// LMRayMarcher_16999480558176
// MI455X (gfx1250) — matrix operand order repaired and checked by reading, not compiled, not run on hardware
//
#include <hip/hip_runtime.h>

typedef __attribute__((ext_vector_type(16))) _Float16 v16h;
typedef __attribute__((ext_vector_type(8)))  float    v8f;

#define N_ITER   20
#define ONE_PLUS_LAMB 1.001f
#define LOG2E    1.4426950408889634f
#define LN2      0.6931471805599453f

// One wave (32 lanes) processes a tile of 16 rays.
// Lane L owns A-matrix row (L%16) and the 32-element K-slice dictated by the
// 16-bit 16x32 WMMA A layout: K = 16*(i/8) + base + (i%8), base = (L<16)?0:8.
// B = W2 broadcast across all 16 columns (column-independent -> layout-safe).
__global__ __launch_bounds__(256) void lm_ray_march(
    const float* __restrict__ x, const float* __restrict__ u,
    const float* __restrict__ W1, const float* __restrict__ b1,
    const float* __restrict__ W2, const float* __restrict__ b2,
    float* __restrict__ out, int n)
{
    __shared__ float sW1[3 * 64];
    __shared__ float sb1[64];
    __shared__ float sW2[64];
    __shared__ float sb2;
    __shared__ float sY[8][16];   // per-wave y exchange
    __shared__ float sDY[8][16];  // per-wave dy exchange

    // Stage weights in LDS once per block.
    for (int i = threadIdx.x; i < 192; i += 256) sW1[i] = W1[i];
    for (int i = threadIdx.x; i < 64;  i += 256) { sb1[i] = b1[i]; sW2[i] = W2[i]; }
    if (threadIdx.x == 0) sb2 = b2[0];
    __syncthreads();

    const int lane = threadIdx.x & 31;
    const int wid  = threadIdx.x >> 5;
    const int tile = blockIdx.x * 8 + wid;
    const int r    = lane & 15;             // ray-in-tile (A-matrix row)
    const int ray  = tile * 16 + r;
    const int base = (lane < 16) ? 0 : 8;   // K-slice offset per half-wave

    const float x0 = x[ray * 3 + 0], x1 = x[ray * 3 + 1], x2 = x[ray * 3 + 2];
    const float u0 = u[ray * 3 + 0], u1 = u[ray * 3 + 1], u2 = u[ray * 3 + 2];

    // Precompute a = x@W1 + b1, c = u@W1 for this lane's 32 K values (once).
    float a[32], c[32];
    #pragma unroll
    for (int i = 0; i < 32; ++i) {
        const int K  = ((i >> 3) << 4) + base + (i & 7);
        const float w0 = sW1[K], w1 = sW1[64 + K], w2 = sW1[128 + K];
        a[i] = fmaf(x0, w0, fmaf(x1, w1, fmaf(x2, w2, sb1[K])));
        c[i] = fmaf(u0, w0, fmaf(u1, w1, u2 * w2));
    }

    // B operands: W2 replicated across the 16 N-columns (constant over iters).
    const int kb = (lane < 16) ? 0 : 16;
    v16h B0, B1;
    #pragma unroll
    for (int j = 0; j < 16; ++j) {
        B0[j] = (_Float16)sW2[((j >> 3) << 4) + base + (j & 7)];
        B1[j] = (_Float16)sW2[32 + ((j >> 3) << 4) + base + (j & 7)];
    }

    const float b2v    = sb2;
    const int rowbase  = (lane >> 1) & 8;   // lane 0 -> rows 0..7, lane 16 -> rows 8..15
    float d = 0.0f;

    for (int it = 0; it < N_ITER; ++it) {
        v16h A0, A1, D0, D1;   // h and dh in f16 for the WMMA reduction
        #pragma unroll
        for (int i = 0; i < 32; ++i) {
            float pre = fmaf(d, c[i], a[i]);
            pre = fminf(fmaxf(pre, -30.0f), 30.0f);
            const float e  = __builtin_amdgcn_exp2f(pre * LOG2E);
            const float t  = 1.0f + e;
            const float rt = __builtin_amdgcn_rcpf(t);
            const float h  = __builtin_amdgcn_logf(t) * LN2;   // softplus
            const float dh = (e * rt) * c[i];                  // sigmoid * c
            if (i < 16) { A0[i]      = (_Float16)h; D0[i]      = (_Float16)dh; }
            else        { A1[i - 16] = (_Float16)h; D1[i - 16] = (_Float16)dh; }
        }

        // y = h @ W2, dy = dh @ W2 : K=64 via two 16x16x32 f16 WMMAs each,
        // f32 accumulation in the C operand.  EXEC is all-1s here (required).
        v8f Y = {}; v8f DY = {};
        Y  = __builtin_amdgcn_wmma_f32_16x16x32_f16(false, A0, false, B0, (short)0, Y,  false, false);
        Y  = __builtin_amdgcn_wmma_f32_16x16x32_f16(false, A1, false, B1, (short)0, Y,  false, false);
        DY = __builtin_amdgcn_wmma_f32_16x16x32_f16(false, D0, false, B0, (short)0, DY, false, false);
        DY = __builtin_amdgcn_wmma_f32_16x16x32_f16(false, D1, false, B1, (short)0, DY, false, false);

        // D layout: VGPR v holds row v (lanes 0-15) / row v+8 (lanes 16-31);
        // all 16 columns identical.  Lanes 0 and 16 publish the 16 per-ray
        // scalars through LDS in a single divergent region.
        if ((lane & 15) == 0) {
            #pragma unroll
            for (int v = 0; v < 8; ++v) {
                sY[wid][rowbase + v]  = Y[v];
                sDY[wid][rowbase + v] = DY[v];
            }
        }
        // Same-wave DS ops are in-order; no barrier needed.
        const float y  = sY[wid][r] + b2v;
        const float dy = sDY[wid][r];
        d -= y * __builtin_amdgcn_rcpf(ONE_PLUS_LAMB * dy);
    }

    if (lane < 16 && ray < n) out[ray] = d;
}

extern "C" void kernel_launch(void* const* d_in, const int* in_sizes, int n_in,
                              void* d_out, int out_size, void* d_ws, size_t ws_size,
                              hipStream_t stream) {
    const float* x  = (const float*)d_in[0];
    const float* u  = (const float*)d_in[1];
    const float* W1 = (const float*)d_in[2];
    const float* b1 = (const float*)d_in[3];
    const float* W2 = (const float*)d_in[4];
    const float* b2 = (const float*)d_in[5];
    float* out = (float*)d_out;

    const int n      = in_sizes[0] / 3;        // number of rays
    const int tiles  = (n + 15) / 16;          // 16 rays per wave
    const int blocks = (tiles + 7) / 8;        // 8 waves per 256-thread block
    lm_ray_march<<<blocks, 256, 0, stream>>>(x, u, W1, b1, W2, b2, out, n);
}
